// LinearMultiHeadAttention_44495861187339
// MI455X (gfx1250) — matrix operand order repaired and checked by reading, not compiled, not run on hardware
//
#include <hip/hip_runtime.h>
#include <hip/hip_bf16.h>

// Linear multi-head attention for MI455X (gfx1250), all GEMMs via
// v_wmma_f32_16x16x32_f16 (f16 in, f32 accum).
//
// Shapes: b=4, n=8192, d=512, h=8, c=64.
// ws layout (bytes):
//   Xq16,Xk16,Xv16 : 3 x 32 MiB   (inputs converted to f16)
//   Wq16,Wk16,Wv16 : 3 x 512 KiB  (weights converted to f16)
//   Q16            : 32 MiB       phi(Q), natural [32768][512]
//   KT16,VT16      : 2 x 32 MiB   phi(K)^T and V^T as [b*h][64][8192]
//   kvpart         : 8 MiB        f32 split-K partials [bh*16+split][d][c]
//   kspart         : 512 KiB      f32 ksum partials
//   kvT            : 512 KiB      f32 kv^T [bh][d][c]
//   ksum           : 8 KiB        f32 [bh][c]

typedef __attribute__((ext_vector_type(16))) _Float16 v16h;
typedef __attribute__((ext_vector_type(8)))  _Float16 v8h;
typedef __attribute__((ext_vector_type(8)))  float    v8f;
typedef __attribute__((ext_vector_type(4)))  float    v4f;

#define DM     512
#define SEQ    8192
#define BATCH  4
#define BN     (BATCH*SEQ)   // 32768
#define SPLITS 16
#define EPS_V  1e-4f

static __device__ inline v16h make16(v8h lo, v8h hi) {
  v16h r;
#pragma unroll
  for (int i = 0; i < 8; ++i) { r[i] = lo[i]; r[i + 8] = hi[i]; }
  return r;
}

// A operand (16x32, M rows in lanes): lane L -> row L&15;
// 16 contiguous K starting at (L>>4)*16.
static __device__ inline v16h load_A(const _Float16* rowp, int lane) {
  const int kb = (lane >> 4) << 4;
  v8h lo = *(const v8h*)(rowp + kb);
  v8h hi = *(const v8h*)(rowp + kb + 8);
  return make16(lo, hi);
}

// B operand (32x16, N cols in lanes): lane L -> col L&15;
// 16 contiguous K starting at (L>>4)*16.
static __device__ inline v16h load_B(const _Float16* colp, int lane) {
  const int kb = (lane >> 4) << 4;
  v8h lo = *(const v8h*)(colp + kb);
  v8h hi = *(const v8h*)(colp + kb + 8);
  return make16(lo, hi);
}

static __device__ inline v8f wmma_f16(v16h a, v16h b, v8f c) {
  return __builtin_amdgcn_wmma_f32_16x16x32_f16(false, a, false, b,
                                                (short)0, c, false, false);
}

// ---------------- fp32 -> f16 convert ----------------
__global__ __launch_bounds__(256) void cvt_kernel(const float* __restrict__ src,
                                                  _Float16* __restrict__ dst,
                                                  int n8) {
  int i = blockIdx.x * 256 + threadIdx.x;
  if (i >= n8) return;
  v4f a = ((const v4f*)src)[2 * i];
  v4f b = ((const v4f*)src)[2 * i + 1];
  v8h o;
#pragma unroll
  for (int j = 0; j < 4; ++j) { o[j] = (_Float16)a[j]; o[j + 4] = (_Float16)b[j]; }
  ((v8h*)dst)[i] = o;
}

// ---------------- stage 1: QKV projection + bias + elu+1 ----------------
// grid (BN/128, DM/128, 3): 128(M) x 128(N) tile, 8 waves = 4(M) x 2(N),
// each wave 2 M-tiles x 4 N-tiles (64 acc VGPRs/lane).
__global__ __launch_bounds__(256) void proj_kernel(
    const _Float16* __restrict__ Xq, const _Float16* __restrict__ Xk,
    const _Float16* __restrict__ Xv, const _Float16* __restrict__ Wq,
    const _Float16* __restrict__ Wk, const _Float16* __restrict__ Wv,
    const float* __restrict__ bq, const float* __restrict__ bk,
    const float* __restrict__ bv, _Float16* __restrict__ Q16,
    _Float16* __restrict__ KT, _Float16* __restrict__ VT) {
  const int lane = threadIdx.x & 31;
  const int wid  = threadIdx.x >> 5;
  const int lmn  = lane & 15;
  const int z    = blockIdx.z;
  const _Float16* X = (z == 0) ? Xq : (z == 1) ? Xk : Xv;
  const _Float16* W = (z == 0) ? Wq : (z == 1) ? Wk : Wv;
  const float* bias = (z == 0) ? bq : (z == 1) ? bk : bv;
  const int Mbase = blockIdx.x * 128;
  const int Nbase = blockIdx.y * 128;
  const int wm = wid >> 1, wn = wid & 1;

  v8f acc[2][4];
#pragma unroll
  for (int tm = 0; tm < 2; ++tm)
#pragma unroll
    for (int tn = 0; tn < 4; ++tn) acc[tm][tn] = {};

  for (int kc = 0; kc < DM; kc += 32) {
    v16h A[2];
#pragma unroll
    for (int tm = 0; tm < 2; ++tm) {
      int row = Mbase + wm * 32 + tm * 16 + lmn;
      A[tm] = load_A(X + (size_t)row * DM + kc, lane);
    }
#pragma unroll
    for (int tn = 0; tn < 4; ++tn) {
      int col = Nbase + wn * 64 + tn * 16 + lmn;
      v16h B = load_B(W + (size_t)col * DM + kc, lane);
#pragma unroll
      for (int tm = 0; tm < 2; ++tm) acc[tm][tn] = wmma_f16(A[tm], B, acc[tm][tn]);
    }
  }

  const bool do_phi = (z < 2);
#pragma unroll
  for (int tn = 0; tn < 4; ++tn) {
    int col = Nbase + wn * 64 + tn * 16 + lmn;
    float bb = bias[col];
#pragma unroll
    for (int tm = 0; tm < 2; ++tm) {
      int m0 = Mbase + wm * 32 + tm * 16 + ((lane >> 4) << 3);
      v8h hv;
#pragma unroll
      for (int g = 0; g < 8; ++g) {
        float x = acc[tm][tn][g] + bb;
        if (do_phi) x = (x > 0.f) ? (x + 1.f) : __expf(x);  // elu(x)+1
        hv[g] = (_Float16)x;
      }
      if (z == 0) {  // phi(Q): natural [BN][DM]
#pragma unroll
        for (int g = 0; g < 8; ++g)
          Q16[(size_t)(m0 + g) * DM + col] = hv[g];
      } else {       // phi(K)^T / V^T: [b*8+h][c][m], packed b128 store
        int b = m0 >> 13, ml = m0 & (SEQ - 1);
        int h = col >> 6, c = col & 63;
        _Float16* T = (z == 1) ? KT : VT;
        *(v8h*)(T + ((size_t)(b * 8 + h) * 64 + c) * SEQ + ml) = hv;
      }
    }
  }
}

// ---------------- stage 2: kv split-K partials + ksum partials ----------------
// grid (SPLITS, 32). Output 64x64 = 4x4 tiles; wave w -> c-tile w>>1,
// d-tiles {(w&1)*2, (w&1)*2+1}. Writes f32 partials (no atomics).
__global__ __launch_bounds__(256) void kv_kernel(const _Float16* __restrict__ KT,
                                                 const _Float16* __restrict__ VT,
                                                 float* __restrict__ kvpart,
                                                 float* __restrict__ kspart) {
  const int lane = threadIdx.x & 31;
  const int wid  = threadIdx.x >> 5;
  const int lmn  = lane & 15;
  const int bh = blockIdx.y, split = blockIdx.x;
  const int mbase = split * (SEQ / SPLITS);
  const int ct = wid >> 1, d0 = (wid & 1) * 2;

  v8f acc[2];
#pragma unroll
  for (int j = 0; j < 2; ++j) acc[j] = {};

  for (int kc = 0; kc < SEQ / SPLITS; kc += 32) {
    const _Float16* pa = KT + ((size_t)bh * 64 + ct * 16 + lmn) * SEQ + mbase + kc;
    v16h A = load_A(pa, lane);
#pragma unroll
    for (int j = 0; j < 2; ++j) {
      const _Float16* pb =
          VT + ((size_t)bh * 64 + (d0 + j) * 16 + lmn) * SEQ + mbase + kc;
      v16h B = load_B(pb, lane);
      acc[j] = wmma_f16(A, B, acc[j]);
    }
  }
  float* op = kvpart + (size_t)(bh * SPLITS + split) * 4096;  // [d][c]
#pragma unroll
  for (int j = 0; j < 2; ++j) {
    int dcol = (d0 + j) * 16 + lmn;
#pragma unroll
    for (int g = 0; g < 8; ++g) {
      int crow = ct * 16 + ((lane >> 4) << 3) + g;
      op[dcol * 64 + crow] = acc[j][g];
    }
  }
  // ksum partial: thread t sums 128 m-values of phi(K)^T row c = t&63
  int t = threadIdx.x, c = t & 63, q = t >> 6;
  const _Float16* pk = KT + ((size_t)bh * 64 + c) * SEQ + mbase + q * 128;
  float s = 0.f;
#pragma unroll
  for (int i = 0; i < 16; ++i) {
    v8h vv = *(const v8h*)(pk + i * 8);
#pragma unroll
    for (int j = 0; j < 8; ++j) s += (float)vv[j];
  }
  kspart[(size_t)(bh * SPLITS + split) * 256 + q * 64 + c] = s;
}

// ---------------- stage 2b: reduce partials ----------------
__global__ __launch_bounds__(256) void reduce_kernel(const float* __restrict__ kvpart,
                                                     const float* __restrict__ kspart,
                                                     float* __restrict__ kvT,
                                                     float* __restrict__ ksum) {
  int bh = blockIdx.x, t = threadIdx.x;
  for (int e = t; e < 4096; e += 256) {
    float s = 0.f;
#pragma unroll
    for (int sp = 0; sp < SPLITS; ++sp)
      s += kvpart[(size_t)(bh * SPLITS + sp) * 4096 + e];
    kvT[(size_t)bh * 4096 + e] = s;
  }
  if (t < 64) {
    float s = 0.f;
#pragma unroll
    for (int sp = 0; sp < SPLITS; ++sp)
#pragma unroll
      for (int q = 0; q < 4; ++q)
        s += kspart[(size_t)(bh * SPLITS + sp) * 256 + q * 64 + t];
    ksum[bh * 64 + t] = s;
  }
}

// ---------------- stage 3: out = phi(Q) @ kv, denom via extra WMMA ----------------
// grid (16, 32): block = (bh, 512-row chunk). kv held in regs as f16 B operands.
__global__ __launch_bounds__(256) void out_kernel(const _Float16* __restrict__ Q16,
                                                  const float* __restrict__ kvT,
                                                  const float* __restrict__ ksum,
                                                  float* __restrict__ out) {
  const int lane = threadIdx.x & 31;
  const int wid  = threadIdx.x >> 5;
  const int lmn  = lane & 15;
  const int bh = blockIdx.y, b = bh >> 3, h = bh & 7;
  const int chunk = blockIdx.x;

  v16h Bk[4][2];
#pragma unroll
  for (int tn = 0; tn < 4; ++tn)
#pragma unroll
    for (int ks = 0; ks < 2; ++ks) {
      int d = tn * 16 + lmn;
      int cb = ks * 32 + ((lane >> 4) << 4);
      const float* p = kvT + (size_t)bh * 4096 + d * 64 + cb;
      v8h lo, hi;
#pragma unroll
      for (int i = 0; i < 8; ++i) { lo[i] = (_Float16)p[i]; hi[i] = (_Float16)p[8 + i]; }
      Bk[tn][ks] = make16(lo, hi);
    }
  v16h Bs[2];
#pragma unroll
  for (int ks = 0; ks < 2; ++ks) {
    v8h lo, hi;
#pragma unroll
    for (int i = 0; i < 8; ++i) { lo[i] = (_Float16)0.f; hi[i] = (_Float16)0.f; }
    if (lmn == 0) {
      const float* p = ksum + bh * 64 + ks * 32 + ((lane >> 4) << 4);
#pragma unroll
      for (int i = 0; i < 8; ++i) { lo[i] = (_Float16)p[i]; hi[i] = (_Float16)p[8 + i]; }
    }
    Bs[ks] = make16(lo, hi);
  }

  for (int it = 0; it < 4; ++it) {
    int n0 = chunk * 512 + it * 128 + wid * 16;
    v8f acc[4];
#pragma unroll
    for (int tn = 0; tn < 4; ++tn) acc[tn] = {};
    v8f accd = {};
#pragma unroll
    for (int ks = 0; ks < 2; ++ks) {
      const _Float16* pa =
          Q16 + ((size_t)(b * SEQ + n0 + lmn)) * DM + h * 64 + ks * 32;
      v16h A = load_A(pa, lane);
#pragma unroll
      for (int tn = 0; tn < 4; ++tn) acc[tn] = wmma_f16(A, Bk[tn][ks], acc[tn]);
      accd = wmma_f16(A, Bs[ks], accd);
    }
    // denom lives in lanes 0 (rows 0-7) and 16 (rows 8-15), col 0
    float inv[8];
#pragma unroll
    for (int g = 0; g < 8; ++g)
      inv[g] = 1.0f / (__shfl(accd[g], threadIdx.x & 16, 32) + EPS_V);
    int m0 = n0 + ((lane >> 4) << 3);
#pragma unroll
    for (int tn = 0; tn < 4; ++tn) {
      int col = h * 64 + tn * 16 + lmn;
#pragma unroll
      for (int g = 0; g < 8; ++g)
        out[((size_t)(b * SEQ + m0 + g)) * DM + col] = acc[tn][g] * inv[g];
    }
  }
}

extern "C" void kernel_launch(void* const* d_in, const int* in_sizes, int n_in,
                              void* d_out, int out_size, void* d_ws, size_t ws_size,
                              hipStream_t stream) {
  (void)in_sizes; (void)n_in; (void)out_size; (void)ws_size;
  const float* Xq = (const float*)d_in[0];
  const float* Xk = (const float*)d_in[1];
  const float* Xv = (const float*)d_in[2];
  const float* Wq = (const float*)d_in[3];
  const float* bq = (const float*)d_in[4];
  const float* Wk = (const float*)d_in[5];
  const float* bk = (const float*)d_in[6];
  const float* Wv = (const float*)d_in[7];
  const float* bv = (const float*)d_in[8];
  float* out = (float*)d_out;

  const size_t NX = (size_t)BN * DM;      // 16,777,216 elements
  const size_t NW = (size_t)DM * DM;      // 262,144 elements
  char* p = (char*)d_ws;
  _Float16* Xq16 = (_Float16*)p; p += NX * 2;
  _Float16* Xk16 = (_Float16*)p; p += NX * 2;
  _Float16* Xv16 = (_Float16*)p; p += NX * 2;
  _Float16* Wq16 = (_Float16*)p; p += NW * 2;
  _Float16* Wk16 = (_Float16*)p; p += NW * 2;
  _Float16* Wv16 = (_Float16*)p; p += NW * 2;
  _Float16* Q16  = (_Float16*)p; p += NX * 2;
  _Float16* KT   = (_Float16*)p; p += NX * 2;
  _Float16* VT   = (_Float16*)p; p += NX * 2;
  float* kvpart  = (float*)p;    p += (size_t)32 * SPLITS * 4096 * 4;
  float* kspart  = (float*)p;    p += (size_t)32 * SPLITS * 256 * 4;
  float* kvT     = (float*)p;    p += (size_t)32 * 4096 * 4;
  float* ksum    = (float*)p;    p += (size_t)32 * 64 * 4;

  {
    int n8 = (int)(NX / 8), blk = n8 / 256;
    cvt_kernel<<<blk, 256, 0, stream>>>(Xq, Xq16, n8);
    cvt_kernel<<<blk, 256, 0, stream>>>(Xk, Xk16, n8);
    cvt_kernel<<<blk, 256, 0, stream>>>(Xv, Xv16, n8);
    int n8w = (int)(NW / 8), blkw = n8w / 256;
    cvt_kernel<<<blkw, 256, 0, stream>>>(Wq, Wq16, n8w);
    cvt_kernel<<<blkw, 256, 0, stream>>>(Wk, Wk16, n8w);
    cvt_kernel<<<blkw, 256, 0, stream>>>(Wv, Wv16, n8w);
  }
  proj_kernel<<<dim3(BN / 128, DM / 128, 3), 256, 0, stream>>>(
      Xq16, Xk16, Xv16, Wq16, Wk16, Wv16, bq, bk, bv, Q16, KT, VT);
  kv_kernel<<<dim3(SPLITS, 32), 256, 0, stream>>>(KT, VT, kvpart, kspart);
  reduce_kernel<<<32, 256, 0, stream>>>(kvpart, kspart, kvT, ksum);
  out_kernel<<<dim3(16, 32), 256, 0, stream>>>(Q16, kvT, ksum, out);
}
